// SAGEEncoder_34419867910897
// MI455X (gfx1250) — matrix operand order repaired and checked by reading, not compiled, not run on hardware
//
#include <hip/hip_runtime.h>
#include <hip/hip_bf16.h>

typedef __attribute__((ext_vector_type(16))) __bf16 v16bf;
typedef __attribute__((ext_vector_type(8)))  float  v8f;

#define N_NODES 50000
#define N_EDGES 800000
#define D_IN    128
#define D_HID   256
#define D_EMB   128

// ---------------- workspace layout (bytes) ----------------
constexpr size_t wsAlign(size_t x) { return (x + 255) & ~size_t(255); }
constexpr size_t OFF_AGGSUM = 0;                                              // float [N,128] (25.6 MB)
constexpr size_t OFF_DEG    = OFF_AGGSUM + (size_t)N_NODES * D_IN * 4;        // float [N]
constexpr size_t OFF_XB     = wsAlign(OFF_DEG + (size_t)N_NODES * 4);         // bf16 [N,128]
constexpr size_t OFF_AGGB   = wsAlign(OFF_XB + (size_t)N_NODES * D_IN * 2);   // bf16 [N,128]
constexpr size_t OFF_H2     = wsAlign(OFF_AGGB + (size_t)N_NODES * D_IN * 2); // bf16 [N,256] (25.6 MB)
constexpr size_t OFF_W      = wsAlign(OFF_H2 + (size_t)N_NODES * D_HID * 2);  // bf16 weights (320 KB)
constexpr size_t OFF_H1     = OFF_AGGSUM;  // bf16 [N,256] aliases agg_sum (dead by then)

// ---------------- kernels ----------------
__global__ void k_zero(float* __restrict__ p, long long n) {
  long long i = (long long)blockIdx.x * blockDim.x + threadIdx.x;
  long long stride = (long long)gridDim.x * blockDim.x;
  for (; i < n; i += stride) p[i] = 0.0f;
}

// one wave per edge; each lane handles 4 consecutive features
__global__ void k_scatter(const float* __restrict__ x, const int* __restrict__ ei,
                          float* __restrict__ aggsum, float* __restrict__ deg) {
  int t = blockIdx.x * blockDim.x + threadIdx.x;
  int e = t >> 5;
  if (e >= N_EDGES) return;
  int lane = t & 31;
  int src = ei[e];
  int dst = ei[N_EDGES + e];
  const float4 v = *reinterpret_cast<const float4*>(x + (size_t)src * D_IN + lane * 4);
  float* out = aggsum + (size_t)dst * D_IN + lane * 4;
  atomicAdd(out + 0, v.x);
  atomicAdd(out + 1, v.y);
  atomicAdd(out + 2, v.z);
  atomicAdd(out + 3, v.w);
  if (lane == 0) atomicAdd(deg + dst, 1.0f);
}

// agg = agg_sum / max(deg,1); convert agg and x to bf16
__global__ void k_finalize(const float* __restrict__ x, const float* __restrict__ aggsum,
                           const float* __restrict__ deg,
                           __bf16* __restrict__ xb, __bf16* __restrict__ aggb) {
  int t = blockIdx.x * blockDim.x + threadIdx.x;
  int node = t >> 5;
  if (node >= N_NODES) return;
  int lane = t & 31;
  float inv = 1.0f / fmaxf(deg[node], 1.0f);
  size_t base = (size_t)node * D_IN + lane * 4;
  float4 a  = *reinterpret_cast<const float4*>(aggsum + base);
  float4 xv = *reinterpret_cast<const float4*>(x + base);
  aggb[base + 0] = (__bf16)(a.x * inv);
  aggb[base + 1] = (__bf16)(a.y * inv);
  aggb[base + 2] = (__bf16)(a.z * inv);
  aggb[base + 3] = (__bf16)(a.w * inv);
  xb[base + 0] = (__bf16)xv.x;
  xb[base + 1] = (__bf16)xv.y;
  xb[base + 2] = (__bf16)xv.z;
  xb[base + 3] = (__bf16)xv.w;
}

// Store W[k][n] (row-major K x N, fp32) into bf16 arena pre-swizzled in WMMA
// B-fragment order: frag index = (n/16)*(K/32) + k/32, and within the 512-element
// fragment: lane = ((k%32)/16)*16 + n%16 holds element e = k%16 at [lane*16 + e].
// This lets the GEMM read each lane's B fragment as one aligned 32-byte v16bf.
__device__ __forceinline__ void cvt_store_swz(const float* __restrict__ W,
                                              __bf16* __restrict__ dst,
                                              int i, int K, int N) {
  int k = i / N, n = i - k * N;
  int kb = k >> 5, kr = k & 31;
  int lane = ((kr >> 4) << 4) | (n & 15);
  int e = kr & 15;
  size_t frag = (size_t)(n >> 4) * (K >> 5) + kb;
  dst[frag * 512 + lane * 16 + e] = (__bf16)W[i];
}

__global__ void k_cvt_w(const float* __restrict__ Wl, const float* __restrict__ Wr,
                        const float* __restrict__ Wm1, const float* __restrict__ Wm2,
                        __bf16* __restrict__ wb) {
  int i = blockIdx.x * blockDim.x + threadIdx.x;
  if (i < 32768)        cvt_store_swz(Wl,  wb,           i,          128, 256);
  else if (i < 65536)   cvt_store_swz(Wr,  wb + 32768,   i - 32768,  128, 256);
  else if (i < 131072)  cvt_store_swz(Wm1, wb + 65536,   i - 65536,  256, 256);
  else if (i < 163840)  cvt_store_swz(Wm2, wb + 131072,  i - 131072, 256, 128);
}

// Wave-tile WMMA GEMM: out[M, NCOL] = epilogue( A0[M,KDIM]@B0 (+ A1@B1) + bias )
// A: row-major [M, KDIM] bf16. B: pre-swizzled fragment arena (see cvt_store_swz).
// Each wave computes a 16 x (16*NT) tile with NT v8f accumulators.
template <int NT, int KDIM, int NCOL, bool RELU, bool DUAL, bool OUTF32>
__global__ void k_gemm(const __bf16* __restrict__ A0, const __bf16* __restrict__ B0,
                       const __bf16* __restrict__ A1, const __bf16* __restrict__ B1,
                       const float* __restrict__ bias,
                       __bf16* __restrict__ outb, float* __restrict__ outf) {
  constexpr int MT   = N_NODES / 16;        // 3125 row tiles
  constexpr int NGRP = NCOL / (16 * NT);    // column groups per row
  constexpr int KB   = KDIM >> 5;           // K fragments per column tile
  int wid = blockIdx.x * (blockDim.x >> 5) + (threadIdx.x >> 5);  // wave-uniform
  if (wid >= MT * NGRP) return;             // whole-wave exit -> EXEC stays all-ones
  int mtile = wid / NGRP;
  int ngrp  = wid % NGRP;
  int mrow  = mtile * 16;
  int nbase = ngrp * (16 * NT);
  int lane = threadIdx.x & 31;
  int g  = lane >> 4;    // lane half
  int li = lane & 15;

  v8f acc[NT] = {};

  for (int pass = 0; pass < (DUAL ? 2 : 1); ++pass) {
    const __bf16* __restrict__ A = pass ? A1 : A0;
    const __bf16* __restrict__ B = pass ? B1 : B0;
    for (int kb = 0; kb < KB; ++kb) {
      // A fragment: row = mrow+li; per lane one contiguous 16-element chunk:
      // K in [g*16, g*16+16)
      const __bf16* Arow = A + (size_t)(mrow + li) * KDIM + kb * 32;
      v16bf a;
#pragma unroll
      for (int v = 0; v < 8; ++v) {
        int bk = g * 16 + 2 * v;
        a[2 * v]     = Arow[bk];
        a[2 * v + 1] = Arow[bk + 1];
      }
#pragma unroll
      for (int t = 0; t < NT; ++t) {
        // swizzled B: one aligned 32B load per lane
        size_t frag = (size_t)((nbase >> 4) + t) * KB + kb;
        v16bf b = *reinterpret_cast<const v16bf*>(B + frag * 512 + lane * 16);
        acc[t] = __builtin_amdgcn_wmma_f32_16x16x32_bf16(
            false, a, false, b, (short)0, acc[t], false, false);
      }
    }
  }

  // epilogue: C/D layout -> element r of v8f is row M = g*8 + r, col N = li
#pragma unroll
  for (int t = 0; t < NT; ++t) {
    int nc = nbase + t * 16 + li;
    float bv = bias[nc];
#pragma unroll
    for (int r = 0; r < 8; ++r) {
      float v = acc[t][r] + bv;
      if (RELU) v = fmaxf(v, 0.0f);
      size_t o = (size_t)(mrow + g * 8 + r) * NCOL + nc;
      if (OUTF32) outf[o] = v;
      else        outb[o] = (__bf16)v;
    }
  }
}

// ---------------- launcher ----------------
extern "C" void kernel_launch(void* const* d_in, const int* in_sizes, int n_in,
                              void* d_out, int out_size, void* d_ws, size_t ws_size,
                              hipStream_t stream) {
  const float* x   = (const float*)d_in[0];
  const int*   ei  = (const int*)d_in[1];
  const float* W_l = (const float*)d_in[2];
  const float* b_l = (const float*)d_in[3];
  const float* W_r = (const float*)d_in[4];
  const float* Wm1 = (const float*)d_in[5];
  const float* bm1 = (const float*)d_in[6];
  const float* Wm2 = (const float*)d_in[7];
  const float* bm2 = (const float*)d_in[8];

  char* ws = (char*)d_ws;
  float*  aggsum = (float*)(ws + OFF_AGGSUM);
  float*  deg    = (float*)(ws + OFF_DEG);
  __bf16* xb     = (__bf16*)(ws + OFF_XB);
  __bf16* aggb   = (__bf16*)(ws + OFF_AGGB);
  __bf16* h1     = (__bf16*)(ws + OFF_H1);   // aliases agg_sum (dead after finalize)
  __bf16* h2     = (__bf16*)(ws + OFF_H2);
  __bf16* wb     = (__bf16*)(ws + OFF_W);
  __bf16* wl  = wb;
  __bf16* wr  = wb + 32768;
  __bf16* wm1 = wb + 65536;
  __bf16* wm2 = wb + 131072;

  // 1) zero accumulators (agg_sum and deg are contiguous)
  k_zero<<<2048, 256, 0, stream>>>(aggsum, (long long)N_NODES * D_IN + N_NODES);

  // 2) edge scatter-add (wave per edge)
  k_scatter<<<(N_EDGES * 32) / 256, 256, 0, stream>>>(x, ei, aggsum, deg);

  // 3) mean + bf16 conversion of agg and x
  k_finalize<<<(N_NODES * 32) / 256, 256, 0, stream>>>(x, aggsum, deg, xb, aggb);

  // 4) weights fp32 -> bf16, pre-swizzled into WMMA B-fragment order
  k_cvt_w<<<163840 / 256, 256, 0, stream>>>(W_l, W_r, Wm1, Wm2, wb);

  // 5) h1 = relu(agg@W_l + x@W_r + b_l)   [50000x256], bf16 out
  {
    constexpr int waves = (N_NODES / 16) * (D_HID / 64);  // 12500
    k_gemm<4, 128, 256, true, true, false>
        <<<(waves + 7) / 8, 256, 0, stream>>>(aggb, wl, xb, wr, b_l, h1, nullptr);
  }
  // 6) h2 = relu(h1@Wm1 + bm1)            [50000x256], bf16 out
  {
    constexpr int waves = (N_NODES / 16) * (D_HID / 64);  // 12500
    k_gemm<4, 256, 256, true, false, false>
        <<<(waves + 7) / 8, 256, 0, stream>>>(h1, wm1, nullptr, nullptr, bm1, h2, nullptr);
  }
  // 7) out = h2@Wm2 + bm2                 [50000x128], fp32 out
  {
    constexpr int waves = (N_NODES / 16) * (D_EMB / 64);  // 6250
    k_gemm<4, 256, 128, false, false, true>
        <<<(waves + 7) / 8, 256, 0, stream>>>(h2, wm2, nullptr, nullptr, bm2, nullptr, (float*)d_out);
  }
}
